// VQVAE_14448269984281
// MI455X (gfx1250) — compile-verified, matrix operand order checked by reading, not run on hardware
//
#include <hip/hip_runtime.h>
#include <hip/hip_bf16.h>

typedef __attribute__((ext_vector_type(16))) __bf16 v16bf;
typedef __attribute__((ext_vector_type(8)))  __bf16 v8bf;
typedef __attribute__((ext_vector_type(8)))  float  v8f;

// ---------------------------------------------------------------------------
// Implicit-GEMM convolution via CDNA5 WMMA (wave32), fully shape-specialized.
//   GEMM: M = B*OH*OW output pixels, N = OC, K = IC*KH*KW
// Block = 256 threads = 8 waves, mapped shape-dependently:
//   WN = min(NT,8) waves span N-tiles, WM = 8/WN wave-groups span extra M,
//   so ALL waves run the matrix pipe even when OC is small (enc3: OC=64,
//   dec3: OC=3). Each wave computes four 16x16 WMMA tiles (M sub-tiles),
//   reusing one B (weight) fragment across all four.
// Per 32-wide K step:
//   * 256 threads cooperatively im2col-gather the next (WM*64)x32 bf16 A tile
//     into the alternate LDS buffer (8 elems -> one 16B ds store, per group),
//   * each lane fetches its B fragment as two 16B loads of pre-converted bf16
//     weights (L2-resident, no per-step cvt) + prefetch of the next line,
//   * 4x v_wmma_f32_16x16x32_bf16 accumulate into fp32,
//   * one __syncthreads per step (double-buffered LDS).
// DECONV=1: lhs-dilated stride-2 transposed conv (dilated pos p = o-PAD+kk
// contributes x[p/2] iff p even and in range).
//
// WMMA 16-bit fragment layout (ISA 7.12.2): element i of the 16-elem vector
// maps to k = (i<8 ? i : i+8) + (lane>=16 ? 8 : 0); A row / B col = lane&15;
// C/D: col = lane&15, VGPR r holds row r + 8*(lane>>4).
// ---------------------------------------------------------------------------
template<int IC, int H, int W, int OC, int OH, int OW,
         int KH, int KW, int STRIDE, int PAD, int DECONV, int RELU>
__global__ void __launch_bounds__(256)
conv_wmma(const float* __restrict__ x, const __bf16* __restrict__ w,
          const float* __restrict__ bias, float* __restrict__ y, int Bn)
{
    constexpr int KHW  = KH * KW;
    constexpr int Ktot = IC * KHW;
    constexpr int NT   = (OC + 15) / 16;                    // N wave-tiles
    constexpr int WN   = (NT >= 8) ? 8 : (NT >= 4 ? 4 : (NT >= 2 ? 2 : 1));
    constexpr int WM   = 8 / WN;                            // M wave-groups
    constexpr int NBLK = (NT + WN - 1) / WN;                // N block-tiles
    constexpr int MROWS = WM * 64;                          // rows per block

    const int lane = threadIdx.x & 31;
    const int wave = threadIdx.x >> 5;
    const int half = lane >> 4;
    const int lrow = lane & 15;
    const int mg   = wave / WN;                             // wave M-group
    const int ntw  = wave - mg * WN;                        // wave N index

    const int Mtot    = Bn * OH * OW;
    const int mblocks = (Mtot + MROWS - 1) / MROWS;
    const int mb = blockIdx.x / NBLK;
    const int nb = blockIdx.x - mb * NBLK;
    if (mb >= mblocks) return;
    const int nt = nb * WN + ntw;
    const int n  = nt * 16 + lrow;                          // output channel
    const bool ncol_ok = (nt < NT) && (n < OC);

    __shared__ __bf16 As[2][MROWS][32];                     // double-buffered

    // Cooperative-gather coords: thread -> (WM rows, 8 consecutive k each).
    const int grow = threadIdx.x >> 2;                      // 0..63
    const int gks  = (threadIdx.x & 3) * 8;                 // 0,8,16,24
    int  gb[WM], goy[WM], gox[WM];
    bool gv[WM];
#pragma unroll
    for (int g = 0; g < WM; ++g) {
        const int gm = mb * MROWS + g * 64 + grow;
        gv[g] = (gm < Mtot);
        gb[g] = 0; goy[g] = 0; gox[g] = 0;
        if (gv[g]) {
            gb[g] = gm / (OH * OW);                         // const divisors
            const int rem = gm - gb[g] * (OH * OW);
            goy[g] = rem / OW;
            gox[g] = rem - goy[g] * OW;
        }
    }

    auto gather = [&](int k0, int buf) {
#pragma unroll
        for (int g = 0; g < WM; ++g) {
            v8bf vals;
#pragma unroll
            for (int j = 0; j < 8; ++j) {
                const int k = k0 + gks + j;
                float av = 0.f;
                if (gv[g] && (Ktot % 32 == 0 || k < Ktot)) {
                    const int ic = k / KHW;
                    const int r2 = k - ic * KHW;
                    const int ky = r2 / KW;
                    const int kx = r2 - ky * KW;
                    int iy, ix; bool ok;
                    if (!DECONV) {
                        iy = goy[g] * STRIDE - PAD + ky;
                        ix = gox[g] * STRIDE - PAD + kx;
                        ok = (iy >= 0) && (iy < H) && (ix >= 0) && (ix < W);
                    } else {
                        const int py = goy[g] - PAD + ky;
                        const int px = gox[g] - PAD + kx;
                        iy = py >> 1; ix = px >> 1;
                        ok = (py >= 0) && (px >= 0) && !(py & 1) && !(px & 1)
                             && (iy < H) && (ix < W);
                    }
                    if (ok) av = x[(((size_t)gb[g] * IC + ic) * H + iy) * W + ix];
                }
                vals[j] = (__bf16)av;
            }
            *(v8bf*)&As[buf][g * 64 + grow][gks] = vals;    // one 16B ds store
        }
    };

    v8f acc[4] = {};
    gather(0, 0);
    int p = 0;
    for (int k0 = 0; k0 < Ktot; k0 += 32) {
        __syncthreads();                                    // tile p ready
        if (k0 + 32 < Ktot) gather(k0 + 32, p ^ 1);

        // B fragment: two 16B loads of 8 contiguous bf16 weights each.
        v16bf bfrag;
        if (ncol_ok && (Ktot % 32 == 0 || k0 + 32 <= Ktot)) {
            const v8bf b0 = *(const v8bf*)(w + (size_t)n * Ktot + k0 + half * 8);
            const v8bf b1 = *(const v8bf*)(w + (size_t)n * Ktot + k0 + 16 + half * 8);
#pragma unroll
            for (int i = 0; i < 8; ++i) { bfrag[i] = b0[i]; bfrag[i + 8] = b1[i]; }
            if (k0 + 64 <= Ktot)                            // cover L2 latency
                __builtin_prefetch(w + (size_t)n * Ktot + k0 + 32, 0, 3);
        } else {
#pragma unroll
            for (int i = 0; i < 16; ++i) {
                const int k = k0 + ((i < 8) ? i : i + 8) + half * 8;
                bfrag[i] = (ncol_ok && k < Ktot) ? w[(size_t)n * Ktot + k]
                                                 : (__bf16)0.f;
            }
        }

        // 4 M sub-tiles share the B fragment.
#pragma unroll
        for (int s = 0; s < 4; ++s) {
            v16bf afrag;
            const v8bf lo = *(const v8bf*)&As[p][mg * 64 + s * 16 + lrow][half * 8];
            const v8bf hi = *(const v8bf*)&As[p][mg * 64 + s * 16 + lrow][16 + half * 8];
#pragma unroll
            for (int i = 0; i < 8; ++i) { afrag[i] = lo[i]; afrag[i + 8] = hi[i]; }
            acc[s] = __builtin_amdgcn_wmma_f32_16x16x32_bf16(
                         false, afrag, false, bfrag,
                         (short)0, acc[s], false, false);
        }
        p ^= 1;
    }

    if (ncol_ok) {
        const float bval = bias[n];
#pragma unroll
        for (int s = 0; s < 4; ++s) {
#pragma unroll
            for (int r = 0; r < 8; ++r) {
                const int m = mb * MROWS + mg * 64 + s * 16 + r + half * 8;
                if (m < Mtot) {
                    const int b   = m / (OH * OW);
                    const int rem = m - b * (OH * OW);
                    const int oy  = rem / OW;
                    const int ox  = rem - oy * OW;
                    float v = acc[s][r] + bval;
                    if (RELU) v = fmaxf(v, 0.f);
                    y[(((size_t)b * OC + n) * OH + oy) * OW + ox] = v;
                }
            }
        }
    }
}

// One-time fp32 -> bf16 weight conversion (weights stay L2-resident).
__global__ void __launch_bounds__(256)
cvt_bf16_kernel(const float* __restrict__ src, __bf16* __restrict__ dst, int n)
{
    const int i = blockIdx.x * 256 + threadIdx.x;
    if (i < n) dst[i] = (__bf16)src[i];
}

// ---------------------------------------------------------------------------
// Vector quantization: per pixel, argmin over 512 codes of ||z-c||^2 in D=64,
// write gathered code (straight-through forward value) in NCHW, accumulate
// loss (forward value = (gamma+beta)*mean((q-z)^2) = 1.25*mean).
// ---------------------------------------------------------------------------
__global__ void __launch_bounds__(256)
vq_kernel(const float* __restrict__ z, const float* __restrict__ cb,
          float* __restrict__ vqx, float* __restrict__ loss_out,
          int Bn, int HW, int K)
{
    const int gid = blockIdx.x * blockDim.x + threadIdx.x;
    const int N = Bn * HW;
    if (gid >= N) return;
    const int b   = gid / HW;
    const int pix = gid - b * HW;
    const float* zp = z + (size_t)b * 64 * HW + pix;

    float zf[64];
#pragma unroll
    for (int d = 0; d < 64; ++d) zf[d] = zp[(size_t)d * HW];

    float best = 3.402823466e38f;
    int bestk = 0;
    for (int k = 0; k < K; ++k) {
        const float* c = cb + (size_t)k * 64;
        float dist = 0.f;
#pragma unroll
        for (int d = 0; d < 64; ++d) {
            const float t = zf[d] - c[d];
            dist = fmaf(t, t, dist);
        }
        if (dist < best) { best = dist; bestk = k; }   // first-min = argmin
    }

    const float* c = cb + (size_t)bestk * 64;
    float* vp = vqx + (size_t)b * 64 * HW + pix;
    float part = 0.f;
#pragma unroll
    for (int d = 0; d < 64; ++d) {
        const float q = c[d];
        const float t = q - zf[d];
        part = fmaf(t, t, part);
        vp[(size_t)d * HW] = q;
    }
    atomicAdd(loss_out, part * (1.25f / ((float)N * 64.0f)));
}

__global__ void zero_scalar_kernel(float* p)
{
    if (threadIdx.x == 0 && blockIdx.x == 0) *p = 0.f;
}

static inline int conv_grid(int M, int OC)
{
    const int nt = (OC + 15) / 16;
    const int wn = (nt >= 8) ? 8 : (nt >= 4 ? 4 : (nt >= 2 ? 2 : 1));
    const int wm = 8 / wn;
    const int mblocks = (M + wm * 64 - 1) / (wm * 64);
    const int nblk    = (nt + wn - 1) / wn;
    return mblocks * nblk;
}

extern "C" void kernel_launch(void* const* d_in, const int* in_sizes, int n_in,
                              void* d_out, int out_size, void* d_ws, size_t ws_size,
                              hipStream_t stream)
{
    (void)in_sizes; (void)n_in; (void)ws_size;
    const float* x      = (const float*)d_in[0];
    const float* enc_w1 = (const float*)d_in[1];
    const float* enc_b1 = (const float*)d_in[2];
    const float* enc_w2 = (const float*)d_in[3];
    const float* enc_b2 = (const float*)d_in[4];
    const float* enc_w3 = (const float*)d_in[5];
    const float* enc_b3 = (const float*)d_in[6];
    const float* cb     = (const float*)d_in[7];
    const float* dec_w1 = (const float*)d_in[8];
    const float* dec_b1 = (const float*)d_in[9];
    const float* dec_w2 = (const float*)d_in[10];
    const float* dec_b2 = (const float*)d_in[11];
    const float* dec_w3 = (const float*)d_in[12];
    const float* dec_b3 = (const float*)d_in[13];

    // Workspace layout (bytes), activation buffers reused once dead:
    //   h1 : 128 MiB (reused for dec2 output)   h2 : 64 MiB (reused for dec1)
    //   z  : 16 MiB                             vqx: 16 MiB
    //   then bf16 weight copies (~2.7 MiB total)
    char* wsb = (char*)d_ws;
    float* h1  = (float*)(wsb);
    float* h2  = (float*)(wsb + (size_t)134217728);
    float* z   = (float*)(wsb + (size_t)134217728 + 67108864);
    float* vqx = (float*)(wsb + (size_t)134217728 + 67108864 + 16777216);
    size_t off = (size_t)134217728 + 67108864 + 16777216 + 16777216;

    const int wn1 = 128 * 3 * 16;     // 6144
    const int wn2 = 256 * 128 * 16;   // 524288
    const int wn3 = 64 * 256 * 9;     // 147456
    const int wn4 = 256 * 64 * 9;     // 147456
    const int wn5 = 128 * 256 * 16;   // 524288
    const int wn6 = 3 * 128 * 16;     // 6144
    __bf16* wb1 = (__bf16*)(wsb + off); off += (size_t)wn1 * 2;
    __bf16* wb2 = (__bf16*)(wsb + off); off += (size_t)wn2 * 2;
    __bf16* wb3 = (__bf16*)(wsb + off); off += (size_t)wn3 * 2;
    __bf16* wb4 = (__bf16*)(wsb + off); off += (size_t)wn4 * 2;
    __bf16* wb5 = (__bf16*)(wsb + off); off += (size_t)wn5 * 2;
    __bf16* wb6 = (__bf16*)(wsb + off); off += (size_t)wn6 * 2;

    float* d1  = h2;   // h2 dead after enc3
    float* dd2 = h1;   // h1 dead after enc2
    float* out  = (float*)d_out;
    float* loss = out + (out_size - 1);

    // One-time weight downconversions (tiny; L2-resident thereafter).
    cvt_bf16_kernel<<<(wn1 + 255) / 256, 256, 0, stream>>>(enc_w1, wb1, wn1);
    cvt_bf16_kernel<<<(wn2 + 255) / 256, 256, 0, stream>>>(enc_w2, wb2, wn2);
    cvt_bf16_kernel<<<(wn3 + 255) / 256, 256, 0, stream>>>(enc_w3, wb3, wn3);
    cvt_bf16_kernel<<<(wn4 + 255) / 256, 256, 0, stream>>>(dec_w1, wb4, wn4);
    cvt_bf16_kernel<<<(wn5 + 255) / 256, 256, 0, stream>>>(dec_w2, wb5, wn5);
    cvt_bf16_kernel<<<(wn6 + 255) / 256, 256, 0, stream>>>(dec_w3, wb6, wn6);

    // Encoder: 256 -> 128 -> 64 spatial
    conv_wmma<3, 256, 256, 128, 128, 128, 4, 4, 2, 1, 0, 1>
        <<<conv_grid(16 * 128 * 128, 128), 256, 0, stream>>>(x, wb1, enc_b1, h1, 16);
    conv_wmma<128, 128, 128, 256, 64, 64, 4, 4, 2, 1, 0, 1>
        <<<conv_grid(16 * 64 * 64, 256), 256, 0, stream>>>(h1, wb2, enc_b2, h2, 16);
    conv_wmma<256, 64, 64, 64, 64, 64, 3, 3, 1, 1, 0, 0>
        <<<conv_grid(16 * 64 * 64, 64), 256, 0, stream>>>(h2, wb3, enc_b3, z, 16);

    // Vector quantization (+ loss scalar at the tail of d_out)
    zero_scalar_kernel<<<1, 64, 0, stream>>>(loss);
    vq_kernel<<<(16 * 64 * 64 + 255) / 256, 256, 0, stream>>>(
        z, cb, vqx, loss, 16, 64 * 64, 512);

    // Decoder: 64 -> 128 -> 256 spatial
    conv_wmma<64, 64, 64, 256, 64, 64, 3, 3, 1, 1, 0, 1>
        <<<conv_grid(16 * 64 * 64, 256), 256, 0, stream>>>(vqx, wb4, dec_b1, d1, 16);
    conv_wmma<256, 64, 64, 128, 128, 128, 4, 4, 1, 2, 1, 1>
        <<<conv_grid(16 * 128 * 128, 128), 256, 0, stream>>>(d1, wb5, dec_b2, dd2, 16);
    conv_wmma<128, 128, 128, 3, 256, 256, 4, 4, 1, 2, 1, 0>
        <<<conv_grid(16 * 256 * 256, 3), 256, 0, stream>>>(dd2, wb6, dec_b3, out, 16);
}
